// QuantizationModule_32255204393385
// MI455X (gfx1250) — matrix operand order repaired and checked by reading, not compiled, not run on hardware
//
#include <hip/hip_runtime.h>
#include <hip/hip_bf16.h>
#include <stdint.h>

// NSVQ quantization for MI455X (gfx1250, wave32, WMMA + async LDS DMA).
// N=131072 rows, K=2048 codebooks, D=128.
// - Distance GEMM: bf16 WMMA (f32 accum), 32 rows/wave, B double-buffered in
//   registers (ds loads run ahead of WMMAs), codebook chunks DMA'd to LDS via
//   global_load_async_to_lds_b128 (ASYNCcnt), double-buffered per block.
// - Argmin: index packed into distance mantissa LSBs -> v_min_num_f32 chains.
// - NSVQ epilogue in exact f32; histogram via global atomics.

#define NROWS 131072
#define KCB   2048
#define DIM   128

#define CB_PER_CHUNK 128                       // codebooks per LDS chunk
#define CHUNK_BYTES  (CB_PER_CHUNK * DIM * 2)  // 32 KB (bf16)
#define NCHUNK       (KCB / CB_PER_CHUNK)      // 16
#define TILES_PER_CHUNK (CB_PER_CHUNK / 16)    // 8

typedef __attribute__((ext_vector_type(16))) __bf16 v16bf;
typedef __attribute__((ext_vector_type(8)))  float  v8f;

// ---------------------------------------------------------------------------
// Prep: codebooks f32 -> bf16 (row major), per-codebook |c|^2, zero histogram.
// ---------------------------------------------------------------------------
__global__ void nsvq_prep(const float* __restrict__ cb,
                          __bf16* __restrict__ cb_bf,
                          float* __restrict__ cnorm,
                          int* __restrict__ used)
{
    const int k = blockIdx.x;      // codebook row
    const int t = threadIdx.x;     // 0..127

    float v = cb[(size_t)k * DIM + t];
    cb_bf[(size_t)k * DIM + t] = (__bf16)v;

    __shared__ float red[DIM];
    red[t] = v * v;
    __syncthreads();
    for (int s = DIM / 2; s > 0; s >>= 1) {
        if (t < s) red[t] += red[t + s];
        __syncthreads();
    }
    if (t == 0) {
        cnorm[k] = red[0];
        used[k]  = 0;              // deterministic: reset histogram every launch
    }
}

// Convert one 16x128 f32 row tile into WMMA A-operand bf16 registers.
// 16-bit A layout: lane L holds row M=L&15; kb=0/16 per half-wave;
// v16bf elems 0..15 = K kb..kb+15 (contiguous, matches B).
__device__ __forceinline__ void load_A_tile(const float* __restrict__ xr,
                                            int kb, v16bf A[4])
{
    #pragma unroll
    for (int c = 0; c < 4; ++c) {
        const int k0 = c * 32 + kb;
        float4 f0 = *(const float4*)(xr + k0);
        float4 f1 = *(const float4*)(xr + k0 + 4);
        float4 f2 = *(const float4*)(xr + k0 + 8);
        float4 f3 = *(const float4*)(xr + k0 + 12);
        A[c][0]  = (__bf16)f0.x; A[c][1]  = (__bf16)f0.y;
        A[c][2]  = (__bf16)f0.z; A[c][3]  = (__bf16)f0.w;
        A[c][4]  = (__bf16)f1.x; A[c][5]  = (__bf16)f1.y;
        A[c][6]  = (__bf16)f1.z; A[c][7]  = (__bf16)f1.w;
        A[c][8]  = (__bf16)f2.x; A[c][9]  = (__bf16)f2.y;
        A[c][10] = (__bf16)f2.z; A[c][11] = (__bf16)f2.w;
        A[c][12] = (__bf16)f3.x; A[c][13] = (__bf16)f3.y;
        A[c][14] = (__bf16)f3.z; A[c][15] = (__bf16)f3.w;
    }
}

// ---------------------------------------------------------------------------
// Main: one wave32 per 32-row tile of x. 8 waves/block -> 256 rows/block.
// ---------------------------------------------------------------------------
__global__ void __launch_bounds__(256)
nsvq_main(const float* __restrict__ x,
          const float* __restrict__ cbf32,
          const float* __restrict__ rnd,
          const __bf16* __restrict__ cb_bf,
          const float* __restrict__ cnorm,
          float* __restrict__ out,
          int* __restrict__ used)
{
    extern __shared__ char smem_raw[];         // 2 x CHUNK_BYTES

    const int lane = threadIdx.x & 31;
    const int wave = threadIdx.x >> 5;
    const size_t rowBase = ((size_t)blockIdx.x * 8 + wave) * 32;

    // ---- Async DMA: block cooperatively pulls one 32 KB chunk.
    // 256 threads x 8 x b128 = 32 KB; ASYNCcnt-tracked.
    const uint32_t ldsBase = (uint32_t)(uintptr_t)smem_raw;     // LDS byte offset
    const uint64_t gBase   = (uint64_t)(uintptr_t)cb_bf;
    auto issue_chunk = [&](int chunk, int buf) {
        uint64_t g = gBase + (uint64_t)chunk * CHUNK_BYTES + (uint32_t)(threadIdx.x * 16);
        uint32_t l = ldsBase + (uint32_t)(buf * CHUNK_BYTES) + (uint32_t)(threadIdx.x * 16);
        #pragma unroll
        for (int j = 0; j < CHUNK_BYTES / (256 * 16); ++j) {    // 8 iters
            asm volatile("global_load_async_to_lds_b128 %0, %1, off"
                         :: "v"(l + (uint32_t)(j * 4096)),
                            "v"(g + (uint64_t)(j * 4096))
                         : "memory");
        }
    };

    // ---- A operands: two 16x128 tiles (32 rows), converted once, reused 128x.
    const int m  = lane & 15;
    const int kb = (lane < 16) ? 0 : 16;
    v16bf A0[4], A1[4];
    load_A_tile(x + (rowBase + m) * DIM,      kb, A0);
    load_A_tile(x + (rowBase + 16 + m) * DIM, kb, A1);

    // Running packed argmin (index in low 11 mantissa bits).
    float mv0[8], mv1[8];
    #pragma unroll
    for (int v = 0; v < 8; ++v) { mv0[v] = 3.0e38f; mv1[v] = 3.0e38f; }

    const int n    = lane & 15;                 // C/D column owned by this lane
    const int koff = (lane < 16) ? 0 : 16;      // B half-wave K split

    auto load_B = [&](v16bf Bt[4], const __bf16* sbuf, int t) {
        const __bf16* cbr = sbuf + ((t * 16 + n) * DIM) + koff;
        #pragma unroll
        for (int c = 0; c < 4; ++c)
            Bt[c] = *(const v16bf*)(cbr + c * 32);   // 2x ds_load_b128 each
    };

    // Prologue: pull chunk 0 into buffer 0.
    issue_chunk(0, 0);
    asm volatile("s_wait_asynccnt 0x0" ::: "memory");
    __syncthreads();

    for (int ch = 0; ch < NCHUNK; ++ch) {
        const int buf = ch & 1;
        if (ch + 1 < NCHUNK) issue_chunk(ch + 1, buf ^ 1);   // overlap DMA

        const __bf16* sbuf = (const __bf16*)(smem_raw + buf * CHUNK_BYTES);

        v16bf B[2][4];
        load_B(B[0], sbuf, 0);                  // software pipeline prologue
        float cn_next = cnorm[ch * CB_PER_CHUNK + n];

        #pragma unroll
        for (int t = 0; t < TILES_PER_CHUNK; ++t) {
            const int cur = t & 1;
            const float cn = cn_next;
            if (t + 1 < TILES_PER_CHUNK) {      // run next tile's B ahead
                load_B(B[cur ^ 1], sbuf, t + 1);
                cn_next = cnorm[ch * CB_PER_CHUNK + (t + 1) * 16 + n];
            }

            v8f a0 = {}, a1 = {};
            #pragma unroll
            for (int c = 0; c < 4; ++c)
                a0 = __builtin_amdgcn_wmma_f32_16x16x32_bf16(
                         false, A0[c], false, B[cur][c], (short)0, a0, false, false);
            #pragma unroll
            for (int c = 0; c < 4; ++c)
                a1 = __builtin_amdgcn_wmma_f32_16x16x32_bf16(
                         false, A1[c], false, B[cur][c], (short)0, a1, false, false);

            const uint32_t idx = (uint32_t)(ch * CB_PER_CHUNK + t * 16 + n);
            #pragma unroll
            for (int v = 0; v < 8; ++v) {
                float s = fmaf(-2.0f, a0[v], cn);          // |c|^2 - 2 x.c
                uint32_t u = (__float_as_uint(s) & 0xFFFFF800u) | idx;
                mv0[v] = fminf(mv0[v], __uint_as_float(u));
            }
            #pragma unroll
            for (int v = 0; v < 8; ++v) {
                float s = fmaf(-2.0f, a1[v], cn);
                uint32_t u = (__float_as_uint(s) & 0xFFFFF800u) | idx;
                mv1[v] = fminf(mv1[v], __uint_as_float(u));
            }
        }

        // Own DMA complete + whole block done reading before buffer reuse.
        asm volatile("s_wait_asynccnt 0x0" ::: "memory");
        __syncthreads();
    }

    // Argmin reduction across each 16-lane half (N dimension): pure float min.
    #pragma unroll
    for (int v = 0; v < 8; ++v) {
        #pragma unroll
        for (int mask = 1; mask <= 8; mask <<= 1) {
            mv0[v] = fminf(mv0[v], __shfl_xor(mv0[v], mask, 32));
            mv1[v] = fminf(mv1[v], __shfl_xor(mv1[v], mask, 32));
        }
    }
    // Row r (0..31): value in mv{r<16?0:1}[r&7], lanes ((r>>3)&1)*16 + 0..15.

    // ---- NSVQ epilogue: per row, res_norm/rnd_norm scaling + histogram.
    const int e = lane * 4;                     // 4 f32 elements per lane
    for (int r = 0; r < 32; ++r) {
        const float w  = (r < 16) ? mv0[r & 7] : mv1[r & 7];
        const float wv = __shfl(w, ((r >> 3) & 1) * 16, 32);
        const int idx  = (int)(__float_as_uint(wv) & 0x7FFu);

        const float* xrow = x     + (rowBase + r) * DIM;
        const float* crow = cbf32 + (size_t)idx * DIM;
        const float* rrow = rnd   + (rowBase + r) * DIM;

        const float4 xv = *(const float4*)(xrow + e);
        const float4 cv = *(const float4*)(crow + e);
        const float4 rv = *(const float4*)(rrow + e);

        const float dx0 = xv.x - cv.x, dx1 = xv.y - cv.y;
        const float dx2 = xv.z - cv.z, dx3 = xv.w - cv.w;
        float rss = dx0*dx0 + dx1*dx1 + dx2*dx2 + dx3*dx3;
        float nss = rv.x*rv.x + rv.y*rv.y + rv.z*rv.z + rv.w*rv.w;

        #pragma unroll
        for (int mask = 1; mask <= 16; mask <<= 1) {
            rss += __shfl_xor(rss, mask, 32);
            nss += __shfl_xor(nss, mask, 32);
        }
        const float scale = sqrtf(rss) / (sqrtf(nss) + 1e-12f);

        float4 o;
        o.x = xv.x + scale * rv.x;
        o.y = xv.y + scale * rv.y;
        o.z = xv.z + scale * rv.z;
        o.w = xv.w + scale * rv.w;
        *(float4*)(out + (rowBase + r) * DIM + e) = o;

        if (lane == 0) atomicAdd(&used[idx], 1);
    }
}

// ---------------------------------------------------------------------------
extern "C" void kernel_launch(void* const* d_in, const int* in_sizes, int n_in,
                              void* d_out, int out_size, void* d_ws, size_t ws_size,
                              hipStream_t stream)
{
    const float* x   = (const float*)d_in[0];   // (N, D) f32
    const float* cb  = (const float*)d_in[1];   // (K, D) f32
    const float* rnd = (const float*)d_in[2];   // (N, D) f32

    float* out  = (float*)d_out;                               // (N, D) f32
    int*   used = (int*)(out + (size_t)NROWS * DIM);           // (K,) i32 (bit tail)

    __bf16* cb_bf = (__bf16*)d_ws;                             // K*D bf16 = 512 KB
    float*  cnorm = (float*)((char*)d_ws +
                             (size_t)KCB * DIM * sizeof(__bf16)); // K f32

    nsvq_prep<<<KCB, DIM, 0, stream>>>(cb, cb_bf, cnorm, used);
    nsvq_main<<<NROWS / 256, 256, 2 * CHUNK_BYTES, stream>>>(
        x, cb, rnd, cb_bf, cnorm, out, used);
}
